// TTSModel_37529424233350
// MI455X (gfx1250) — matrix operand order repaired and checked by reading, not compiled, not run on hardware
//
#include <hip/hip_runtime.h>
#include <hip/hip_bf16.h>
#include <math.h>

// TTS model: embedding -> biLSTM encoder -> 900-step attention LSTM decoder -> FC.
// All GEMMs run on v_wmma_f32_16x16x32_bf16 (bf16 operands, fp32 accum).
// Operands are pre-packed into the CDNA5 wave32 WMMA fragment layouts
// (CDNA5 ISA 7.12.2): one fragment = 32 lanes x 8 dwords = 1KB,
// stored lane-major so each lane loads its 8 dwords contiguously.
// Each wave loads one A fragment per K-chunk and feeds TWO WMMAs (two B tiles),
// halving A-side vmem traffic versus one-tile-per-wave.

#define B_    64
#define TT_   128
#define EMB_  256
#define ENC_  512
#define DEC_  1024
#define MEL_  80
#define TMEL_ 900

typedef __attribute__((ext_vector_type(16))) __bf16        bf16x16;
typedef __attribute__((ext_vector_type(8)))  float         floatx8;
typedef __attribute__((ext_vector_type(8)))  unsigned int  uintx8;

__device__ __forceinline__ unsigned short f2bf(float f) {
  unsigned u = __float_as_uint(f);
  u += 0x7fffu + ((u >> 16) & 1u);          // round-to-nearest-even
  return (unsigned short)(u >> 16);
}

__device__ __forceinline__ float sigf(float x) { return 1.f / (1.f + __expf(-x)); }
__device__ __forceinline__ float tanh_fast(float x) {
  float e = __expf(-2.f * x);
  return (1.f - e) / (1.f + e);
}

// --- fragment-layout address helpers -------------------------------------
// A operand (16x32 MxK bf16): lanes 0-15 rows M=0..15 with K groups {0..7},
// {16..23}; lanes 16-31 same rows, K groups {8..15},{24..31}; 2 K per dword.
__device__ __forceinline__ size_t apack_pos(int mtG, int Kc, int row16, int col) {
  int kc = col >> 5, kk = col & 31;
  int grp = kk >> 3;                              // 0..3
  int lanehalf = grp & 1;                         // hi-lanes hold groups 1,3
  int i = ((grp >> 1) << 2) + ((kk & 7) >> 1);    // vgpr index 0..7
  int L = row16 + (lanehalf << 4);
  size_t dw = ((size_t)mtG * Kc + kc) * 256 + (size_t)L * 8 + i;
  return dw * 2 + (kk & 1);
}

__device__ __forceinline__ bf16x16 load_frag(const unsigned* __restrict__ p, int lane) {
  uintx8 v = *(const uintx8*)(p + (size_t)lane * 8);
  return __builtin_bit_cast(bf16x16, v);
}

__device__ __forceinline__ floatx8 wmma_acc(bf16x16 a, bf16x16 b, floatx8 c) {
  return __builtin_amdgcn_wmma_f32_16x16x32_bf16(false, a, false, b, (short)0, c,
                                                 false, false);
}

// --- utility kernels ------------------------------------------------------
__global__ void fill_zero_kernel(float* p, size_t n) {
  for (size_t i = (size_t)blockIdx.x * blockDim.x + threadIdx.x; i < n;
       i += (size_t)gridDim.x * blockDim.x)
    p[i] = 0.f;
}

// Embedding gather fused with A-fragment packing. Rows are time-major:
// row = t*64 + b, cols = EMB. Kc = 8 (K = 256).
__global__ void pack_x_kernel(const int* __restrict__ text,
                              const float* __restrict__ emb,
                              unsigned short* __restrict__ dst) {
  const size_t total = (size_t)B_ * TT_ * EMB_;
  for (size_t idx = (size_t)blockIdx.x * blockDim.x + threadIdx.x; idx < total;
       idx += (size_t)gridDim.x * blockDim.x) {
    int col = (int)(idx & (EMB_ - 1));
    int row = (int)(idx >> 8);          // t*64 + b
    int t = row >> 6, b = row & 63;
    int tok = text[b * TT_ + t];
    float v = emb[(size_t)tok * EMB_ + col];
    dst[apack_pos(row >> 4, 8, row & 15, col)] = f2bf(v);
  }
}

// Pack weight matrices (with row concatenation / zero padding) into B frags.
// B operand (32x16 KxN bf16): lanes 0-15 cols N=0..15 hold K groups {0..7},{16..23}
// (2/dword), lanes 16-31 hold {8..15},{24..31}. Linear dst index == fragment memory order.
// mode 0: encoder  K=768  [Wr(512); Wk(256)]                            N=2048
// mode 1: decoder  K=2176 [dk_ctx(1024); dr(1024); dk_mel(80); pad(48)] N=4096
// mode 2: fc       K=1024 fc_w, N padded 80 -> 96
__global__ void pack_b_kernel(const float* __restrict__ W1,
                              const float* __restrict__ W2,
                              unsigned short* __restrict__ dst,
                              int mode, int Kc, long long total) {
  for (long long idx = (long long)blockIdx.x * blockDim.x + threadIdx.x; idx < total;
       idx += (long long)gridDim.x * blockDim.x) {
    long long r = idx;
    int half = (int)(r & 1); r >>= 1;
    int i = (int)(r & 7);    r >>= 3;
    int L = (int)(r & 31);   r >>= 5;
    int kc = (int)(r % Kc);
    int ntile = (int)(r / Kc);
    int k = kc * 32 + ((i >> 2) << 4) + ((L >> 4) << 3) + ((i & 3) << 1) + half;
    int col = ntile * 16 + (L & 15);
    float v = 0.f;
    if (mode == 0) {
      if (k < 512) v = W1[(size_t)k * 2048 + col];
      else         v = W2[(size_t)(k - 512) * 2048 + col];
    } else if (mode == 1) {
      if      (k < 1024) v = W1[(size_t)k * 4096 + col];
      else if (k < 2048) v = W2[(size_t)(k - 1024) * 4096 + col];
      else if (k < 2128) v = W1[(size_t)(1024 + (k - 2048)) * 4096 + col];
    } else {
      if (k < 1024 && col < MEL_) v = W1[(size_t)k * MEL_ + col];
    }
    dst[idx] = f2bf(v);
  }
}

// --- encoder LSTM step ----------------------------------------------------
// One step of one direction. A = [h(512) | x_t(256)] (K=768, Kc=24),
// B = packed [Wr;Wk] (N=2048). Grid = 32 WGs (16 hidden cols each), 256 thr.
// Wave w: M-tile w&3, gates {w>>2, (w>>2)+2}; one A load feeds two WMMAs.
__global__ void enc_step_kernel(const unsigned* __restrict__ hpack_in,
                                const unsigned* __restrict__ xapack,
                                const unsigned* __restrict__ bpack,
                                const float* __restrict__ bias,
                                float* __restrict__ cbuf,            // 64x512 in-place
                                unsigned short* __restrict__ hpack_out,
                                float* __restrict__ enc_out,         // (B,TT,1024)
                                int t_out, int enc_half, int t_x,
                                float* __restrict__ hstate_out,      // final step only
                                float* __restrict__ cstate_out) {
  __shared__ float zslab[64][65];
  const int tid = threadIdx.x, lane = tid & 31, w = tid >> 5;
  const int n = blockIdx.x;
  const int mt = w & 3, g0 = w >> 2, g1 = g0 + 2;
  floatx8 acc0 = {}, acc1 = {};
  for (int kc = 0; kc < 24; ++kc) {
    const unsigned* Ap = (kc < 16)
        ? hpack_in + ((size_t)(mt * 16 + kc)) * 256
        : xapack + ((size_t)((t_x * 4 + mt) * 8 + (kc - 16))) * 256;
    bf16x16 a  = load_frag(Ap, lane);
    bf16x16 b0 = load_frag(bpack + ((size_t)((g0 * 32 + n) * 24 + kc)) * 256, lane);
    bf16x16 b1 = load_frag(bpack + ((size_t)((g1 * 32 + n) * 24 + kc)) * 256, lane);
    acc0 = wmma_acc(a, b0, acc0);
    acc1 = wmma_acc(a, b1, acc1);
  }
  const int rbase = mt * 16 + ((lane >> 4) << 3);
  const int c0 = g0 * 16 + (lane & 15), c1 = g1 * 16 + (lane & 15);
#pragma unroll
  for (int j = 0; j < 8; ++j) {
    zslab[rbase + j][c0] = acc0[j];
    zslab[rbase + j][c1] = acc1[j];
  }
  __syncthreads();
  for (int u = tid; u < 64 * 16; u += 256) {
    int row = u >> 4, cl = u & 15;
    int colg = n * 16 + cl;
    float zi = zslab[row][cl]      + bias[colg];
    float zf = zslab[row][16 + cl] + bias[ENC_ + colg];
    float zg = zslab[row][32 + cl] + bias[2 * ENC_ + colg];
    float zo = zslab[row][48 + cl] + bias[3 * ENC_ + colg];
    float c  = cbuf[row * ENC_ + colg];
    float c2 = sigf(zf) * c + sigf(zi) * tanh_fast(zg);
    float h2 = sigf(zo) * tanh_fast(c2);
    cbuf[row * ENC_ + colg] = c2;
    enc_out[((size_t)row * TT_ + t_out) * (2 * ENC_) + enc_half * ENC_ + colg] = h2;
    hpack_out[apack_pos(row >> 4, 16, row & 15, colg)] = f2bf(h2);
    if (hstate_out) {
      hstate_out[row * DEC_ + enc_half * ENC_ + colg] = h2;
      cstate_out[row * DEC_ + enc_half * ENC_ + colg] = c2;
    }
  }
}

// --- decoder attention (fp32 VALU, per batch row) -------------------------
// Writes the packed A operand [ctx(1024) | h(1024) | x_t(80) | pad(48)], Kc=68.
__global__ void attn_kernel(const float* __restrict__ h,
                            const float* __restrict__ enc_out,
                            const float* __restrict__ mel,
                            int t,
                            unsigned short* __restrict__ apack_out) {
  __shared__ float hs[DEC_];
  __shared__ float sc[TT_];
  __shared__ float ctx[DEC_];
  __shared__ float sinv;
  const int b = blockIdx.x, tid = threadIdx.x, lane = tid & 31, w = tid >> 5;
  for (int i = tid; i < DEC_; i += 256) hs[i] = h[b * DEC_ + i];
  __syncthreads();
  for (int tp = w; tp < TT_; tp += 8) {
    const float* e = enc_out + ((size_t)b * TT_ + tp) * DEC_;
    float s = 0.f;
    for (int k = lane; k < DEC_; k += 32) s += hs[k] * e[k];
    for (int off = 16; off; off >>= 1) s += __shfl_xor(s, off, 32);
    if (lane == 0) sc[tp] = s;
  }
  __syncthreads();
  if (tid == 0) {
    float m = sc[0];
    for (int i = 1; i < TT_; ++i) m = fmaxf(m, sc[i]);
    float ssum = 0.f;
    for (int i = 0; i < TT_; ++i) { float e = __expf(sc[i] - m); sc[i] = e; ssum += e; }
    sinv = 1.f / ssum;
  }
  __syncthreads();
  for (int d = tid; d < DEC_; d += 256) {
    float acc = 0.f;
    for (int tp = 0; tp < TT_; ++tp)
      acc += sc[tp] * enc_out[((size_t)b * TT_ + tp) * DEC_ + d];
    ctx[d] = acc * sinv;
  }
  __syncthreads();
  const int mtG = b >> 4, r16 = b & 15;
  for (int col = tid; col < 2176; col += 256) {
    float v;
    if (col < 1024)      v = ctx[col];
    else if (col < 2048) v = hs[col - 1024];
    else if (col < 2128) v = (t > 0) ? mel[((size_t)b * TMEL_ + (t - 1)) * MEL_ + (col - 2048)] : 0.f;
    else                 v = 0.f;
    apack_out[apack_pos(mtG, 68, r16, col)] = f2bf(v);
  }
}

// --- decoder LSTM step ----------------------------------------------------
// A = [ctx|h|x_t] (Kc=68), B = packed [dk_ctx;dr;dk_mel;pad] (N=4096).
// Grid = 64 WGs (16 hidden cols each). Wave w: M-tile w&3, gates {w>>2,(w>>2)+2}.
__global__ void dec_step_kernel(const unsigned* __restrict__ apack,
                                const unsigned* __restrict__ bpack,
                                const float* __restrict__ bias,
                                float* __restrict__ cbuf,   // 64x1024 in-place
                                float* __restrict__ hout,   // 64x1024
                                unsigned short* __restrict__ hist_pack,
                                int t) {
  __shared__ float zslab[64][65];
  const int tid = threadIdx.x, lane = tid & 31, w = tid >> 5;
  const int n = blockIdx.x;
  const int mt = w & 3, g0 = w >> 2, g1 = g0 + 2;
  floatx8 acc0 = {}, acc1 = {};
  for (int kc = 0; kc < 68; ++kc) {
    bf16x16 a  = load_frag(apack + ((size_t)(mt * 68 + kc)) * 256, lane);
    bf16x16 b0 = load_frag(bpack + ((size_t)((g0 * 64 + n) * 68 + kc)) * 256, lane);
    bf16x16 b1 = load_frag(bpack + ((size_t)((g1 * 64 + n) * 68 + kc)) * 256, lane);
    acc0 = wmma_acc(a, b0, acc0);
    acc1 = wmma_acc(a, b1, acc1);
  }
  const int rbase = mt * 16 + ((lane >> 4) << 3);
  const int c0 = g0 * 16 + (lane & 15), c1 = g1 * 16 + (lane & 15);
#pragma unroll
  for (int j = 0; j < 8; ++j) {
    zslab[rbase + j][c0] = acc0[j];
    zslab[rbase + j][c1] = acc1[j];
  }
  __syncthreads();
  for (int u = tid; u < 64 * 16; u += 256) {
    int row = u >> 4, cl = u & 15;
    int colg = n * 16 + cl;
    float zi = zslab[row][cl]      + bias[colg];
    float zf = zslab[row][16 + cl] + bias[DEC_ + colg];
    float zg = zslab[row][32 + cl] + bias[2 * DEC_ + colg];
    float zo = zslab[row][48 + cl] + bias[3 * DEC_ + colg];
    float c  = cbuf[row * DEC_ + colg];
    float c2 = sigf(zf) * c + sigf(zi) * tanh_fast(zg);
    float h2 = sigf(zo) * tanh_fast(c2);
    cbuf[row * DEC_ + colg] = c2;
    hout[row * DEC_ + colg] = h2;
    hist_pack[apack_pos(t * 4 + (row >> 4), 32, row & 15, colg)] = f2bf(h2);
  }
}

// --- final FC: mels = h_hist @ fc_w + fc_b, transposed store -------------
// M = 57600 (t*64+b), K = 1024 (Kc=32), N = 80 (6 ntiles, padded).
// Grid (900, 3), 4 waves (128 thr): wave = M-tile, two N-tiles per A load.
__global__ void fc_kernel(const unsigned* __restrict__ hist_pack,
                          const unsigned* __restrict__ bpack,
                          const float* __restrict__ fc_b,
                          float* __restrict__ out) {
  const int tid = threadIdx.x, lane = tid & 31, mt = tid >> 5;
  const int mslab = blockIdx.x;              // time t
  const int nt0 = blockIdx.y * 2, nt1 = nt0 + 1;
  floatx8 acc0 = {}, acc1 = {};
  for (int kc = 0; kc < 32; ++kc) {
    bf16x16 a  = load_frag(hist_pack + ((size_t)((mslab * 4 + mt) * 32 + kc)) * 256, lane);
    bf16x16 b0 = load_frag(bpack + ((size_t)(nt0 * 32 + kc)) * 256, lane);
    bf16x16 b1 = load_frag(bpack + ((size_t)(nt1 * 32 + kc)) * 256, lane);
    acc0 = wmma_acc(a, b0, acc0);
    acc1 = wmma_acc(a, b1, acc1);
  }
  const int b = mt * 16 + ((lane >> 4) << 3);   // batch row base
  const int col0 = nt0 * 16 + (lane & 15), col1 = col0 + 16;
  if (col0 < MEL_) {
    const float bv = fc_b[col0];
#pragma unroll
    for (int j = 0; j < 8; ++j)
      out[((size_t)(b + j) * TMEL_ + mslab) * MEL_ + col0] = acc0[j] + bv;
  }
  if (col1 < MEL_) {
    const float bv = fc_b[col1];
#pragma unroll
    for (int j = 0; j < 8; ++j)
      out[((size_t)(b + j) * TMEL_ + mslab) * MEL_ + col1] = acc1[j] + bv;
  }
}

// --- host orchestration ---------------------------------------------------
extern "C" void kernel_launch(void* const* d_in, const int* in_sizes, int n_in,
                              void* d_out, int out_size, void* d_ws, size_t ws_size,
                              hipStream_t stream) {
  (void)in_sizes; (void)n_in; (void)out_size; (void)ws_size;
  const int*   text = (const int*)d_in[0];
  const float* mel  = (const float*)d_in[1];
  const float* emb  = (const float*)d_in[2];
  const float* ef_k = (const float*)d_in[3];
  const float* ef_r = (const float*)d_in[4];
  const float* ef_b = (const float*)d_in[5];
  const float* eb_k = (const float*)d_in[6];
  const float* eb_r = (const float*)d_in[7];
  const float* eb_b = (const float*)d_in[8];
  const float* dk   = (const float*)d_in[9];
  const float* dr   = (const float*)d_in[10];
  const float* db   = (const float*)d_in[11];
  const float* fc_w = (const float*)d_in[12];
  const float* fc_b = (const float*)d_in[13];
  float* out = (float*)d_out;

  char* ws = (char*)d_ws;
  size_t off = 0;
  auto alloc = [&](size_t bytes) -> char* {
    char* p = ws + off;
    off += (bytes + 255) & ~(size_t)255;
    return p;
  };
  // fragment regions: <n frags> * 256 dwords * 4B
  unsigned short* xapack = (unsigned short*)alloc((size_t)512 * 8 * 1024);   // 4 MB
  unsigned short* encbf  = (unsigned short*)alloc((size_t)128 * 24 * 1024);  // 3 MB
  unsigned short* encbb  = (unsigned short*)alloc((size_t)128 * 24 * 1024);  // 3 MB
  unsigned short* decb   = (unsigned short*)alloc((size_t)256 * 68 * 1024);  // 17 MB
  unsigned short* fcb    = (unsigned short*)alloc((size_t)6 * 32 * 1024);
  unsigned short* hpackA = (unsigned short*)alloc((size_t)4 * 16 * 1024);
  unsigned short* hpackB = (unsigned short*)alloc((size_t)4 * 16 * 1024);
  float* enc_c   = (float*)alloc((size_t)B_ * ENC_ * 4);
  float* enc_out = (float*)alloc((size_t)B_ * TT_ * 2 * ENC_ * 4);           // 32 MB
  float* dec_h   = (float*)alloc((size_t)B_ * DEC_ * 4);
  float* dec_c   = (float*)alloc((size_t)B_ * DEC_ * 4);
  unsigned short* ctxh = (unsigned short*)alloc((size_t)4 * 68 * 1024);
  unsigned short* hist = (unsigned short*)alloc((size_t)3600 * 32 * 1024);   // 118 MB

  // one-time (per call) packing of activation-independent data
  pack_x_kernel<<<1024, 256, 0, stream>>>(text, emb, xapack);
  pack_b_kernel<<<1024, 256, 0, stream>>>(ef_r, ef_k, encbf, 0, 24, (long long)128 * 24 * 512);
  pack_b_kernel<<<1024, 256, 0, stream>>>(eb_r, eb_k, encbb, 0, 24, (long long)128 * 24 * 512);
  pack_b_kernel<<<2048, 256, 0, stream>>>(dk, dr, decb, 1, 68, (long long)256 * 68 * 512);
  pack_b_kernel<<<128, 256, 0, stream>>>(fc_w, nullptr, fcb, 2, 32, (long long)6 * 32 * 512);

  // encoder forward
  fill_zero_kernel<<<64, 256, 0, stream>>>((float*)hpackA, (size_t)4 * 16 * 256);
  fill_zero_kernel<<<64, 256, 0, stream>>>(enc_c, (size_t)B_ * ENC_);
  for (int s = 0; s < TT_; ++s) {
    const unsigned* hin = (const unsigned*)((s & 1) ? hpackB : hpackA);
    unsigned short* hout = (s & 1) ? hpackA : hpackB;
    bool last = (s == TT_ - 1);
    enc_step_kernel<<<32, 256, 0, stream>>>(hin, (const unsigned*)xapack,
        (const unsigned*)encbf, ef_b, enc_c, hout, enc_out,
        /*t_out=*/s, /*half=*/0, /*t_x=*/s,
        last ? dec_h : nullptr, last ? dec_c : nullptr);
  }
  // encoder backward
  fill_zero_kernel<<<64, 256, 0, stream>>>((float*)hpackA, (size_t)4 * 16 * 256);
  fill_zero_kernel<<<64, 256, 0, stream>>>(enc_c, (size_t)B_ * ENC_);
  for (int s = 0; s < TT_; ++s) {
    const unsigned* hin = (const unsigned*)((s & 1) ? hpackB : hpackA);
    unsigned short* hout = (s & 1) ? hpackA : hpackB;
    bool last = (s == TT_ - 1);
    int tx = TT_ - 1 - s;
    enc_step_kernel<<<32, 256, 0, stream>>>(hin, (const unsigned*)xapack,
        (const unsigned*)encbb, eb_b, enc_c, hout, enc_out,
        /*t_out=*/tx, /*half=*/1, /*t_x=*/tx,
        last ? dec_h : nullptr, last ? dec_c : nullptr);
  }

  // decoder: 900 sequential (attention, LSTM) pairs
  for (int t = 0; t < TMEL_; ++t) {
    attn_kernel<<<64, 256, 0, stream>>>(dec_h, enc_out, mel, t, ctxh);
    dec_step_kernel<<<64, 256, 0, stream>>>((const unsigned*)ctxh,
        (const unsigned*)decb, db, dec_c, dec_h, hist, t);
  }

  // final FC over the whole h history, with (t,b)->(b,t) transpose in store
  dim3 fg(TMEL_, 3);
  fc_kernel<<<fg, 128, 0, stream>>>((const unsigned*)hist, (const unsigned*)fcb,
                                    fc_b, out);
}
